// RNN_20134806684353
// MI455X (gfx1250) — matrix operand order repaired and checked by reading, not compiled, not run on hardware
//
#include <hip/hip_runtime.h>
#include <hip/hip_bf16.h>
#include <math.h>

// ---------------- problem constants ----------------
constexpr int SEQ   = 1024;
constexpr int BATCH = 128;
constexpr int IDIM  = 300;   // input dim
constexpr int HID   = 256;   // hidden
constexpr int NC    = 5;     // classes
constexpr int KPAD  = 320;   // input dim padded to 10 k-tiles of 32
// LDS row strides (bf16 elems): multiple of 8 -> rows 16B-aligned (ds_load_b128),
// and (stride/2) mod 64 chosen so lanes 0..15 hit distinct banks.
constexpr int WSTRIDE_A = 328; // kernel A: 656B/row = 164 dw; 36*m mod 64 distinct
constexpr int HSTRIDE   = 264; // kernel B: 528B/row = 132 dw;  4*m mod 64 distinct

// ---------------- WMMA fragment types ----------------
typedef __attribute__((ext_vector_type(16))) __bf16       bf16x16;
typedef __attribute__((ext_vector_type(8)))  float        f32x8;
typedef __attribute__((ext_vector_type(4)))  unsigned int u32x4;

union Frag {
    u32x4   q[2];   // two 16B halves -> ds_load_b128 each
    bf16x16 v;
};

__device__ inline float fast_tanh(float x) {
#if __has_builtin(__builtin_amdgcn_tanhf)
    return __builtin_amdgcn_tanhf(x);   // V_TANH_F32 (gfx1250 TRANS op)
#else
    return tanhf(x);
#endif
}

// A-fragment (16x32 bf16, MxK), lane = half*16 + m:
//   VGPRs 0..3 hold k = 8*half + {0..7}, VGPRs 4..7 hold k = 8*half + {16..23}
//   -> two contiguous 16B runs at k0 = 8*half and k0+16.
__device__ inline void load_afrag(Frag& f, const __bf16* row, int kbase, int half) {
    const __bf16* p = row + kbase + 8 * half;
    f.q[0] = *(const u32x4*)(p);
    f.q[1] = *(const u32x4*)(p + 16);
}
// B-fragment (32x16 bf16, KxN), lane = half*16 + n:
//   VGPRs 0..3 hold k = 8*half + {0..7}, VGPRs 4..7 hold k = 8*half + {16..23}
//   -> two contiguous 16B runs at k0 = 8*half and k0+16 (same as A).
__device__ inline void load_bfrag(Frag& f, const __bf16* row, int kbase, int half) {
    const __bf16* p = row + kbase + 8 * half;
    f.q[0] = *(const u32x4*)(p);
    f.q[1] = *(const u32x4*)(p + 16);
}

// =====================================================================
// Kernel A: P[s*B+b, h] = X @ W_ih^T + (b_ih + b_hh), bf16 WMMA, fp32 out
// grid: (SEQ*BATCH)/64 blocks of 512 threads; block tile = 64 x 256
// =====================================================================
__global__ void __launch_bounds__(512)
rnn_xproj_wmma(const float* __restrict__ X, const float* __restrict__ W_ih,
               const float* __restrict__ b_ih, const float* __restrict__ b_hh,
               float* __restrict__ P)
{
    extern __shared__ __bf16 smem[];
    __bf16* Wl = smem;                         // [HID][WSTRIDE_A]  W_ih bf16, zero-padded K
    __bf16* Xl = smem + HID * WSTRIDE_A;       // [64][WSTRIDE_A]   X tile bf16

    const int tid = threadIdx.x;
    const long rowBase = (long)blockIdx.x * 64;

    for (int idx = tid; idx < HID * WSTRIDE_A; idx += 512) {
        int n = idx / WSTRIDE_A, k = idx - n * WSTRIDE_A;
        float v = (k < IDIM) ? W_ih[n * IDIM + k] : 0.0f;
        Wl[idx] = (__bf16)v;
    }
    for (int idx = tid; idx < 64 * WSTRIDE_A; idx += 512) {
        int r = idx / WSTRIDE_A, k = idx - r * WSTRIDE_A;
        float v = (k < IDIM) ? X[(rowBase + r) * IDIM + k] : 0.0f;
        Xl[idx] = (__bf16)v;
    }
    __syncthreads();

    const int wave = tid >> 5;          // 0..15 -> N tile (16 cols each)
    const int lane = tid & 31;
    const int half = lane >> 4;
    const int l16  = lane & 15;
    const int nGlb = wave * 16 + l16;

    const float bias = b_ih[nGlb] + b_hh[nGlb];

    const f32x8 vzero = {0.f, 0.f, 0.f, 0.f, 0.f, 0.f, 0.f, 0.f};
    f32x8 acc[4];
    acc[0] = vzero; acc[1] = vzero; acc[2] = vzero; acc[3] = vzero;

    const __bf16* wrow = Wl + nGlb * WSTRIDE_A;
    for (int j = 0; j < KPAD / 32; ++j) {
        Frag bf;
        load_bfrag(bf, wrow, 32 * j, half);
        #pragma unroll
        for (int mt = 0; mt < 4; ++mt) {
            Frag af;
            load_afrag(af, Xl + (mt * 16 + l16) * WSTRIDE_A, 32 * j, half);
            acc[mt] = __builtin_amdgcn_wmma_f32_16x16x32_bf16(
                false, af.v, false, bf.v, (short)0, acc[mt], false, false);
        }
    }

    #pragma unroll
    for (int mt = 0; mt < 4; ++mt) {
        #pragma unroll
        for (int r = 0; r < 8; ++r) {
            long row = rowBase + mt * 16 + r + 8 * half;   // C/D layout: m = r + 8*half
            P[row * HID + nGlb] = acc[mt][r] + bias;
        }
    }
}

// =====================================================================
// Kernel B: 1024-step tanh RNN scan + hoisted output projection.
// grid: 8 blocks x 512 threads; block owns batch rows [wg*16, wg*16+16).
// Wave w holds its 256x16 W_hh column block as 8 persistent B-fragments.
// h kept in double-buffered LDS (bf16); Sum_t h_t kept in fp32 registers.
// =====================================================================
__global__ void __launch_bounds__(512)
rnn_scan_wmma(const float* __restrict__ P, const float* __restrict__ W_hh,
              const float* __restrict__ W_out, const float* __restrict__ b_out,
              float* __restrict__ out)
{
    extern __shared__ __bf16 smem[];
    __bf16* Whh  = smem;                               // [HID][HSTRIDE]  B-layout: Whh[n][k]
    __bf16* hbuf = smem + HID * HSTRIDE;               // [2][16][HSTRIDE] h double buffer
    float*  yacc = (float*)(hbuf + 2 * 16 * HSTRIDE);  // [16][NC]

    const int tid  = threadIdx.x;
    const int wg   = blockIdx.x;            // batch chunk
    const int wave = tid >> 5, lane = tid & 31;
    const int half = lane >> 4, l16 = lane & 15;
    const int nGlb = wave * 16 + l16;       // hidden column owned by this lane

    for (int idx = tid; idx < HID * HSTRIDE; idx += 512) {
        int n = idx / HSTRIDE, k = idx - n * HSTRIDE;
        float v = (k < HID) ? W_hh[n * HID + k] : 0.0f;  // B[k][n] = W_hh[n][k]
        Whh[idx] = (__bf16)v;
    }
    for (int idx = tid; idx < 2 * 16 * HSTRIDE; idx += 512) hbuf[idx] = (__bf16)0.0f;
    for (int idx = tid; idx < 16 * NC; idx += 512) yacc[idx] = 0.0f;
    __syncthreads();

    // Persistent B fragments: 8 k-tiles x 8 VGPRs of W_hh (register resident)
    Frag bfr[8];
    #pragma unroll
    for (int j = 0; j < 8; ++j)
        load_bfrag(bfr[j], Whh + nGlb * HSTRIDE, 32 * j, half);

    const f32x8 vzero = {0.f, 0.f, 0.f, 0.f, 0.f, 0.f, 0.f, 0.f};
    f32x8 hsum = vzero;

    const float* Pbase = P + ((long)wg * 16) * HID + nGlb;

    for (int t = 0; t < SEQ; ++t) {
        const __bf16* hread  = hbuf + (t & 1) * 16 * HSTRIDE + l16 * HSTRIDE;
        __bf16*       hwrite = hbuf + ((t + 1) & 1) * 16 * HSTRIDE;

        // Two independent accumulate chains (even/odd k-tiles) for WMMA overlap.
        f32x8 acc0 = vzero, acc1 = vzero;
        #pragma unroll
        for (int j = 0; j < 8; j += 2) {
            Frag af0, af1;
            load_afrag(af0, hread, 32 * j,      half);
            load_afrag(af1, hread, 32 * (j+1),  half);
            acc0 = __builtin_amdgcn_wmma_f32_16x16x32_bf16(
                false, af0.v, false, bfr[j].v,   (short)0, acc0, false, false);
            acc1 = __builtin_amdgcn_wmma_f32_16x16x32_bf16(
                false, af1.v, false, bfr[j+1].v, (short)0, acc1, false, false);
        }

        const float* Pt = Pbase + (long)t * BATCH * HID;  // L2-resident (128MB < 192MB L2)
        if (t + 1 < SEQ)
            __builtin_prefetch(Pbase + (long)(t + 1) * BATCH * HID, 0, 1);

        #pragma unroll
        for (int r = 0; r < 8; ++r) {
            int m = r + 8 * half;                                 // C/D layout row
            float h = fast_tanh(acc0[r] + acc1[r] + Pt[m * HID]); // tanh(xp + h @ W_hh^T)
            hsum[r] += h;                                         // fp32 running sum of h
            hwrite[m * HSTRIDE + nGlb] = (__bf16)h;               // publish h_{t+1} (bf16)
        }
        __syncthreads();   // one barrier/step thanks to double buffering
    }

    // Hoisted output projection: y[m][c] = sum_n hsum[m][n] * W_out[c][n]
    float wo[NC];
    #pragma unroll
    for (int c = 0; c < NC; ++c) wo[c] = W_out[c * HID + nGlb];
    #pragma unroll
    for (int r = 0; r < 8; ++r) {
        int m = r + 8 * half;
        #pragma unroll
        for (int c = 0; c < NC; ++c)
            atomicAdd(&yacc[m * NC + c], hsum[r] * wo[c]);   // ds_add_f32
    }
    __syncthreads();

    if (tid < 16) {
        int m = tid;
        float v[NC], mx = -INFINITY;
        #pragma unroll
        for (int c = 0; c < NC; ++c) {
            v[c] = yacc[m * NC + c] + (float)SEQ * b_out[c];
            mx = fmaxf(mx, v[c]);
        }
        float s = 0.f;
        #pragma unroll
        for (int c = 0; c < NC; ++c) s += __expf(v[c] - mx);
        float lse = mx + __logf(s);
        #pragma unroll
        for (int c = 0; c < NC; ++c)
            out[(wg * 16 + m) * NC + c] = v[c] - lse;
    }
}

// =====================================================================
extern "C" void kernel_launch(void* const* d_in, const int* in_sizes, int n_in,
                              void* d_out, int out_size, void* d_ws, size_t ws_size,
                              hipStream_t stream)
{
    const float* X     = (const float*)d_in[0];  // [1024,128,300]
    const float* W_ih  = (const float*)d_in[1];  // [256,300]
    const float* W_hh  = (const float*)d_in[2];  // [256,256]
    const float* b_ih  = (const float*)d_in[3];  // [256]
    const float* b_hh  = (const float*)d_in[4];  // [256]
    const float* W_out = (const float*)d_in[5];  // [5,256]
    const float* b_out = (const float*)d_in[6];  // [5]
    float* out = (float*)d_out;                  // [128,5]

    float* P = (float*)d_ws;                     // x_proj: [SEQ*BATCH, HID] fp32 = 128 MB

    const size_t shA = (size_t)(HID + 64) * WSTRIDE_A * sizeof(__bf16);           // ~205 KB
    rnn_xproj_wmma<<<(SEQ * BATCH) / 64, 512, shA, stream>>>(X, W_ih, b_ih, b_hh, P);

    const size_t shB = (size_t)(HID * HSTRIDE + 2 * 16 * HSTRIDE) * sizeof(__bf16)
                     + (size_t)16 * NC * sizeof(float);                           // ~149 KB
    rnn_scan_wmma<<<BATCH / 16, 512, shB, stream>>>(P, W_hh, W_out, b_out, out);
}
